// FlashAttention_91319594647562
// MI455X (gfx1250) — matrix operand order repaired and checked by reading, not compiled, not run on hardware
//
#include <hip/hip_runtime.h>

typedef __attribute__((ext_vector_type(16))) _Float16 v16h;
typedef __attribute__((ext_vector_type(8)))  float    v8f;
typedef __attribute__((ext_vector_type(4)))  int      v4i;

#define BB 2
#define NN 2048
#define MM 2048
#define HH 16
#define DD 64
#define DMODEL (HH * DD)

#if __has_builtin(__builtin_amdgcn_global_load_async_to_lds_b128) && \
    __has_builtin(__builtin_amdgcn_s_wait_asynccnt)
#define USE_ASYNC_LDS 1
#else
#define USE_ASYNC_LDS 0
#endif

#if USE_ASYNC_LDS
// Builtin signature (probe-confirmed): (global int4*, lds int4*, imm, imm).
#define GAS1(p) ((__attribute__((address_space(1))) v4i*)(void*)(p))
#define LAS3(p) ((__attribute__((address_space(3))) v4i*)(void*)(p))
#endif

__device__ __forceinline__ v8f wmma_f16(v16h a, v16h b, v8f c) {
  return __builtin_amdgcn_wmma_f32_16x16x32_f16(false, a, false, b, (short)0, c,
                                                false, false);
}

// ---------------------------------------------------------------------------
// Kernel 0a: one-shot convert Q/K [B,N,H,D] f32 -> head-major [B,H,N,D] f16.
// ---------------------------------------------------------------------------
__global__ __launch_bounds__(256) void cvt_qk_kernel(
    const float* __restrict__ Q, const float* __restrict__ K,
    _Float16* __restrict__ Qh, _Float16* __restrict__ Kh) {
  const int idx = blockIdx.x * 256 + threadIdx.x;  // [B,H,N,D] linear (dest)
  const int d = idx & (DD - 1);
  int r = idx >> 6;
  const int n = r & (NN - 1);
  r >>= 11;
  const int h = r & (HH - 1);
  const int b = r >> 4;
  const int src = ((b * NN + n) * HH + h) * DD + d;
  Qh[idx] = (_Float16)Q[src];
  Kh[idx] = (_Float16)K[src];
}

// ---------------------------------------------------------------------------
// Kernel 0b: one-shot convert V [B,N,H,D] f32 -> d-major [B,H,D,M] f16.
// LDS tile transpose keeps both global sides coalesced.
// ---------------------------------------------------------------------------
__global__ __launch_bounds__(256) void cvt_v_kernel(const float* __restrict__ V,
                                                    _Float16* __restrict__ Vt) {
  __shared__ _Float16 tile[DD][DD + 8];  // 64x72 halves, padded
  const int mb = blockIdx.x % (MM / 64);
  const int bh = blockIdx.x / (MM / 64);
  const int h = bh % HH, b = bh / HH;

  // load: 4 threads per m-row, 16 contiguous d each (coalesced f32 reads)
  {
    const int tm = threadIdx.x >> 2;
    const int td = (threadIdx.x & 3) * 16;
    const float* src =
        V + ((size_t)(b * NN + mb * 64 + tm) * HH + h) * DD + td;
#pragma unroll
    for (int j = 0; j < 16; ++j) tile[td + j][tm] = (_Float16)src[j];
  }
  __syncthreads();
  // store: 4 threads per d-row, 16 contiguous m each (coalesced f16 writes)
  {
    const int sd = threadIdx.x >> 2;
    const int sm = (threadIdx.x & 3) * 16;
    _Float16* dst =
        Vt + ((size_t)(b * HH + h) * DD + sd) * MM + mb * 64 + sm;
#pragma unroll
    for (int j = 0; j < 16; ++j) dst[j] = tile[sd][sm + j];
  }
}

// ---------------------------------------------------------------------------
// Kernel 0c: one-shot convert W (row-major, 1024x1024) f32 -> f16.
// ---------------------------------------------------------------------------
__global__ __launch_bounds__(256) void cvt_w_kernel(const float* __restrict__ W,
                                                    _Float16* __restrict__ Wh) {
  const int idx = blockIdx.x * 256 + threadIdx.x;
  Wh[idx] = (_Float16)W[idx];
}

// ---------------------------------------------------------------------------
// Kernel 1: flash attention. Block = 8 waves = 128 queries of one (b,h).
// K (key-major) and V (d-major) 32-key tiles staged in LDS, double-buffered,
// async prefetch of next tile overlapped with WMMAs on current.
// ---------------------------------------------------------------------------
__global__ __launch_bounds__(256) void fa_kernel(
    const _Float16* __restrict__ Qh, const _Float16* __restrict__ Kh,
    const _Float16* __restrict__ Vt, const int* __restrict__ causal_p,
    _Float16* __restrict__ X) {
  __shared__ __align__(16) _Float16 ldsK[2][32][DD];   // key-major, 2x4KB
  __shared__ __align__(16) _Float16 ldsVt[2][DD][32];  // d-major,   2x4KB
  __shared__ __align__(16) _Float16 ldsP[8][16][32];   // per-wave P relayout

  const int tid = threadIdx.x;
  const int lane = tid & 31, wave = tid >> 5;
  const int lh = lane >> 4;  // 16-lane group
  const int lm = lane & 15;

  const int nQB = NN / 128;
  const int qBlock = blockIdx.x % nQB;
  const int t = blockIdx.x / nQB;
  const int h = t % HH, b = t / HH;
  const int qBase0 = qBlock * 128;
  const int qBase = qBase0 + wave * 16;
  const int causal = causal_p[0];
  const float scale = 0.125f;  // 1/sqrt(64)

  const _Float16* kslice = Kh + ((size_t)(b * HH + h)) * MM * DD;
  const _Float16* vslice = Vt + ((size_t)(b * HH + h)) * DD * MM;

  // ---- Q A-fragments (f16 head-major: fully contiguous per row) ----------
  const _Float16* qrow = Qh + (((size_t)(b * HH + h)) * NN + (qBase + lm)) * DD;
  v16h aQ[2];
#pragma unroll
  for (int c = 0; c < 2; ++c) {
#pragma unroll
    for (int i = 0; i < 8; ++i) {
      aQ[c][i]     = qrow[c * 32 + lh * 8 + i];
      aQ[c][8 + i] = qrow[c * 32 + 16 + lh * 8 + i];
    }
  }

  float mrow[8], lrow[8];
  v8f o[4] = {};
#pragma unroll
  for (int i = 0; i < 8; ++i) { mrow[i] = -INFINITY; lrow[i] = 0.0f; }

  const int waveKEnd = causal ? (qBase + 15) : (MM - 1);
  const int blkKEnd  = causal ? (qBase0 + 127) : (MM - 1);  // block-uniform

  // Cooperative tile copy: 256 threads x 16B per tile.
  // K tile: rows contiguous (key-major). V tile: 64 d-rows x 32 keys.
  auto load_tiles = [&](int buf, int kb) {
    const _Float16* ks = kslice + (size_t)kb * DD + tid * 8;
    const _Float16* vs = vslice + (size_t)(tid >> 2) * MM + kb + (tid & 3) * 8;
    _Float16* kd = &ldsK[buf][0][0] + tid * 8;
    _Float16* vd = &ldsVt[buf][tid >> 2][(tid & 3) * 8];
#if USE_ASYNC_LDS
    __builtin_amdgcn_global_load_async_to_lds_b128(GAS1(ks), LAS3(kd), 0, 0);
    __builtin_amdgcn_global_load_async_to_lds_b128(GAS1(vs), LAS3(vd), 0, 0);
#else
    *(uint4*)kd = *(const uint4*)ks;
    *(uint4*)vd = *(const uint4*)vs;
#endif
  };

  load_tiles(0, 0);
#if USE_ASYNC_LDS
  __builtin_amdgcn_s_wait_asynccnt(0);
#endif
  __syncthreads();

  for (int kb = 0; kb <= blkKEnd; kb += 32) {
    const int cur = (kb >> 5) & 1;
    if (kb + 32 <= blkKEnd) load_tiles(cur ^ 1, kb + 32);  // prefetch next

    if (kb <= waveKEnd) {  // wave-uniform predicate (EXEC stays full)
      // ---- scores: S = Q K^T, two 16-key subtiles, K=32 chunks over D ----
      v8f s[2];
#pragma unroll
      for (int st = 0; st < 2; ++st) {
        v8f acc = {};
#pragma unroll
        for (int c = 0; c < 2; ++c) {
          const _Float16* kr = &ldsK[cur][st * 16 + lm][c * 32 + lh * 8];
          v16h bk;
#pragma unroll
          for (int j = 0; j < 16; ++j) bk[j] = kr[j < 8 ? j : j + 8];
          acc = wmma_f16(aQ[c], bk, acc);
        }
        s[st] = acc;
      }

      // ---- scale (+ causal mask only on diagonal-straddling blocks) ----
      const bool needMask = (causal != 0) && (kb + 31 > qBase);
      float rm[8];
      if (needMask) {
#pragma unroll
        for (int i = 0; i < 8; ++i) {
          const int qi = qBase + lh * 8 + i;
          float v0 = s[0][i] * scale;
          float v1 = s[1][i] * scale;
          if (kb + lm > qi)      v0 = -INFINITY;
          if (kb + 16 + lm > qi) v1 = -INFINITY;
          s[0][i] = v0;
          s[1][i] = v1;
          rm[i] = fmaxf(v0, v1);
        }
      } else {
#pragma unroll
        for (int i = 0; i < 8; ++i) {
          s[0][i] *= scale;
          s[1][i] *= scale;
          rm[i] = fmaxf(s[0][i], s[1][i]);
        }
      }
#pragma unroll
      for (int i = 0; i < 8; ++i) {
        float v = rm[i];
#pragma unroll
        for (int off = 1; off < 16; off <<= 1)
          v = fmaxf(v, __shfl_xor(v, off, 16));
        rm[i] = v;
      }

      // ---- online softmax update ---------------------------------------
      float fac[8];
#pragma unroll
      for (int i = 0; i < 8; ++i) {
        const float mn = fmaxf(mrow[i], rm[i]);
        fac[i] = __expf(mrow[i] - mn);
        mrow[i] = mn;
        s[0][i] = __expf(s[0][i] - mn);
        s[1][i] = __expf(s[1][i] - mn);
      }
#pragma unroll
      for (int i = 0; i < 8; ++i) {
        float v = s[0][i] + s[1][i];
#pragma unroll
        for (int off = 1; off < 16; off <<= 1) v += __shfl_xor(v, off, 16);
        lrow[i] = lrow[i] * fac[i] + v;
      }
#pragma unroll
      for (int g = 0; g < 4; ++g)
#pragma unroll
        for (int i = 0; i < 8; ++i) o[g][i] *= fac[i];

      // ---- relayout P: C/D fragment -> LDS -> A fragment ---------------
      asm volatile("s_wait_dscnt 0" ::: "memory");
#pragma unroll
      for (int i = 0; i < 8; ++i) {
        ldsP[wave][lh * 8 + i][lm]      = (_Float16)s[0][i];
        ldsP[wave][lh * 8 + i][16 + lm] = (_Float16)s[1][i];
      }
      asm volatile("s_wait_dscnt 0" ::: "memory");
      v16h aP;
      {
        const _Float16* rowp = &ldsP[wave][lm][0];
#pragma unroll
        for (int i = 0; i < 8; ++i) {
          aP[i]     = rowp[lh * 8 + i];
          aP[8 + i] = rowp[16 + lh * 8 + i];
        }
      }

      // ---- O += P * V (V d-major in LDS: contiguous B-fragment reads) --
#pragma unroll
      for (int g = 0; g < 4; ++g) {
        const _Float16* vr = &ldsVt[cur][g * 16 + lm][lh * 8];
        v16h bv;
#pragma unroll
        for (int j = 0; j < 16; ++j) bv[j] = vr[j < 8 ? j : j + 8];
        o[g] = wmma_f16(aP, bv, o[g]);
      }
    }

#if USE_ASYNC_LDS
    __builtin_amdgcn_s_wait_asynccnt(0);  // next tile landed in LDS
#endif
    __syncthreads();
  }

  // ---- normalize and emit X[b*N + q][h*64 + d] as f16 --------------------
#pragma unroll
  for (int i = 0; i < 8; ++i) lrow[i] = 1.0f / lrow[i];
#pragma unroll
  for (int g = 0; g < 4; ++g) {
#pragma unroll
    for (int i = 0; i < 8; ++i) {
      const int q = qBase + lh * 8 + i;
      X[(size_t)(b * NN + q) * DMODEL + h * DD + g * 16 + lm] =
          (_Float16)(o[g][i] * lrow[i]);
    }
  }
}

// ---------------------------------------------------------------------------
// Kernel 2: Y = X @ W^T + b (f16 inputs, f32 out). Wave = 16 rows x 64 cols.
// ---------------------------------------------------------------------------
__global__ __launch_bounds__(256) void proj_kernel(
    const _Float16* __restrict__ X, const _Float16* __restrict__ Wh,
    const float* __restrict__ bias, float* __restrict__ Y) {
  const int lane = threadIdx.x & 31;
  const int wave = threadIdx.x >> 5;
  const int lh = lane >> 4, lm = lane & 15;

  const int tile = blockIdx.x * 8 + wave;  // 4096 tiles
  const int rowTile = tile >> 4;
  const int colGrp = tile & 15;
  const int row0 = rowTile * 16;
  const int col0 = colGrp * 64;

  v8f acc[4] = {};
  const _Float16* xrow = X + (size_t)(row0 + lm) * DMODEL;
  for (int kc = 0; kc < DMODEL; kc += 32) {
    v16h a;
#pragma unroll
    for (int i = 0; i < 8; ++i) {
      a[i]     = xrow[kc + lh * 8 + i];
      a[8 + i] = xrow[kc + 16 + lh * 8 + i];
    }
#pragma unroll
    for (int g = 0; g < 4; ++g) {
      // B[k,c] = W[c,k]: W rows contiguous in k -> per-lane two 8-half runs.
      const _Float16* wrow =
          Wh + (size_t)(col0 + g * 16 + lm) * DMODEL + kc + lh * 8;
      v16h bw;
#pragma unroll
      for (int j = 0; j < 16; ++j) bw[j] = wrow[j < 8 ? j : j + 8];
      acc[g] = wmma_f16(a, bw, acc[g]);
    }
  }
#pragma unroll
  for (int g = 0; g < 4; ++g) {
    const float bb = bias[col0 + g * 16 + lm];
#pragma unroll
    for (int i = 0; i < 8; ++i) {
      Y[(size_t)(row0 + lh * 8 + i) * DMODEL + col0 + g * 16 + lm] =
          acc[g][i] + bb;
    }
  }
}

extern "C" void kernel_launch(void* const* d_in, const int* in_sizes, int n_in,
                              void* d_out, int out_size, void* d_ws,
                              size_t ws_size, hipStream_t stream) {
  const float* Q      = (const float*)d_in[0];
  const float* K      = (const float*)d_in[1];
  const float* V      = (const float*)d_in[2];
  const float* W      = (const float*)d_in[3];
  const float* bO     = (const float*)d_in[4];
  const int*   causal = (const int*)d_in[5];
  float*       Y      = (float*)d_out;

  const size_t nQKV = (size_t)BB * HH * NN * DD;  // 4.19M elems
  _Float16* ws16 = (_Float16*)d_ws;
  _Float16* Qh = ws16;                 // 8 MB, [B,H,N,D]
  _Float16* Kh = ws16 + nQKV;          // 8 MB, [B,H,N,D]
  _Float16* Vt = ws16 + 2 * nQKV;      // 8 MB, [B,H,D,M]
  _Float16* X  = ws16 + 3 * nQKV;      // 8 MB, [B*N, DMODEL]
  _Float16* Wh = ws16 + 4 * nQKV;      // 2 MB

  dim3 blk(256);
  cvt_qk_kernel<<<(int)(nQKV / 256), blk, 0, stream>>>(Q, K, Qh, Kh);
  cvt_v_kernel<<<BB * HH * (MM / 64), blk, 0, stream>>>(V, Vt);
  cvt_w_kernel<<<(DMODEL * DMODEL) / 256, blk, 0, stream>>>(W, Wh);
  fa_kernel<<<BB * HH * (NN / 128), blk, 0, stream>>>(Qh, Kh, Vt, causal, X);
  proj_kernel<<<(BB * NN / 16) * (DMODEL / 64) / 8, blk, 0, stream>>>(X, Wh, bO,
                                                                      Y);
}
